// T_NN_48936857370836
// MI455X (gfx1250) — matrix operand order repaired and checked by reading, not compiled, not run on hardware
//
#include <hip/hip_runtime.h>

// Problem constants (from reference)
#define BDIM   64
#define NDIM   512
#define MDIM   30
#define E2D    4
#define NE     32
#define NE4    8
#define NF     64
#define PAIRS  (BDIM * NDIM)   // 32768
#define GROUPS (PAIRS / 16)    // 2048 (16-row groups for fit MLP)

typedef __attribute__((ext_vector_type(16))) _Float16 v16h;
typedef __attribute__((ext_vector_type(8)))  float    v8f;

// Fast reciprocal: single v_rcp_f32 (~1 ULP) instead of the IEEE divide
// expansion (div_scale/rcp/fma Newton chain). Plenty for f16 WMMA operands.
__device__ __forceinline__ float fast_rcp(float x) {
  return __builtin_amdgcn_rcpf(x);
}
__device__ __forceinline__ float sigmoid_f(float x) {
  // 1/(1+e^-x); x->-inf: rcp(inf)=0 -> 0 ; x->+inf -> 1
  return fast_rcp(1.0f + __expf(-x));
}
__device__ __forceinline__ float tanh_f(float x) {
  // tanh(x) = 1 - 2/(e^{2x}+1); saturates correctly at +-inf with fast rcp
  return 1.0f - 2.0f * fast_rcp(__expf(2.0f * x) + 1.0f);
}

__device__ __forceinline__ v8f wmma16x16x32(v16h a, v16h b, v8f c) {
  // D = A(16x32 f16) * B(32x16 f16) + C(16x16 f32)
  return __builtin_amdgcn_wmma_f32_16x16x32_f16(
      /*neg_a=*/false, a, /*neg_b=*/false, b,
      /*c_mod=*/(short)0, c, /*reuse_a=*/false, /*reuse_b=*/false);
}

// Gather an A fragment (16x32 f16, ISA layout) from a row-major f16 buffer.
// lane L: M = L&15 ; VGPR v holds K = (v&4 ? 16:0) + (L<16 ? 0:8) + (v&3)*2 (+1)
__device__ __forceinline__ v16h load_a_frag(const _Float16* sm, int lane, int ld) {
  int m  = lane & 15;
  int kh = (lane < 16) ? 0 : 8;
  v16h a;
#pragma unroll
  for (int v = 0; v < 8; ++v) {
    int kb = ((v & 4) ? 16 : 0) + kh + (v & 3) * 2;
    a[2 * v]     = sm[m * ld + kb];
    a[2 * v + 1] = sm[m * ld + kb + 1];
  }
  return a;
}

// ---------------------------------------------------------------------------
// k0: zero per-b accumulators
__global__ void k0_zero(float* acc) {
  if (threadIdx.x < BDIM) acc[threadIdx.x] = 0.0f;
}

// ---------------------------------------------------------------------------
// k_pack: pack fit-MLP weights into WMMA B-fragment layout (f16) in workspace.
// B fragment (32x16): lane L: N = t*16 + (L&15); K half by (L>=16); 2 f16/VGPR.
__global__ void __launch_bounds__(256)
k_pack(const float* __restrict__ Wf0, const float* __restrict__ Wf1,
       const float* __restrict__ Wf2,
       _Float16* __restrict__ pBf0, _Float16* __restrict__ pBf1,
       _Float16* __restrict__ pBf2) {
  int tid = blockIdx.x * 256 + threadIdx.x;
  int nt  = gridDim.x * 256;
  // Wf0: [256][64] -> 8 k-steps x 4 n-tiles x 512
  for (int e = tid; e < 8 * 4 * 512; e += nt) {
    int frag = e >> 9, r = e & 511;
    int s = frag >> 2, t = frag & 3;
    int flane = r >> 4, i = r & 15;
    int v = i >> 1, h = i & 1;
    int n = t * 16 + (flane & 15);
    int k = s * 32 + ((v & 4) ? 16 : 0) + ((flane >= 16) ? 8 : 0) + (v & 3) * 2 + h;
    pBf0[e] = (_Float16)Wf0[k * NF + n];
  }
  // Wf1 / Wf2: [64][64] -> 2 k-steps x 4 n-tiles x 512
  for (int e = tid; e < 2 * 4 * 512; e += nt) {
    int frag = e >> 9, r = e & 511;
    int s = frag >> 2, t = frag & 3;
    int flane = r >> 4, i = r & 15;
    int v = i >> 1, h = i & 1;
    int n = t * 16 + (flane & 15);
    int k = s * 32 + ((v & 4) ? 16 : 0) + ((flane >= 16) ? 8 : 0) + (v & 3) * 2 + h;
    pBf1[e] = (_Float16)Wf1[k * NF + n];
    pBf2[e] = (_Float16)Wf2[k * NF + n];
  }
}

// ---------------------------------------------------------------------------
// k1: embedding MLP (layer0 VALU; layers 1-2 WMMA) + small contractions.
// One (b,n) pair per wave; 8 waves / block; D stored to ws in A-fragment order.
__global__ void __launch_bounds__(256)
k1_emb(const float* __restrict__ Sg, const float* __restrict__ R,
       const float* __restrict__ We0, const float* __restrict__ be0,
       const float* __restrict__ We1, const float* __restrict__ be1,
       const float* __restrict__ We2, const float* __restrict__ be2,
       _Float16* __restrict__ wsD) {
  __shared__ float    sWe0[E2D * NE];
  __shared__ float    sbe0[NE], sbe1[NE], sbe2[NE];
  __shared__ _Float16 sB1[2 * 512];          // We1 B-fragments (t = 0,1)
  __shared__ _Float16 sB2[2 * 512];          // We2 B-fragments
  __shared__ float4   sR4[8][32];
  __shared__ _Float16 sHa[8][32 * NE];       // stage A (h0, then G)
  __shared__ _Float16 sHb[8][32 * NE];       // stage B (h1)
  __shared__ float    sA8[8][32];

  int tid = threadIdx.x;
  if (tid < E2D * NE) sWe0[tid] = We0[tid];
  if (tid < NE) { sbe0[tid] = be0[tid]; sbe1[tid] = be1[tid]; sbe2[tid] = be2[tid]; }
  for (int e = tid; e < 1024; e += 256) {
    int t = e >> 9, r = e & 511;
    int flane = r >> 4, i = r & 15;
    int v = i >> 1, h = i & 1;
    int n = t * 16 + (flane & 15);
    int k = ((v & 4) ? 16 : 0) + ((flane >= 16) ? 8 : 0) + (v & 3) * 2 + h;
    sB1[e] = (_Float16)We1[k * NE + n];
    sB2[e] = (_Float16)We2[k * NE + n];
  }
  __syncthreads();

  int lane = tid & 31, w = tid >> 5;
  int pair = blockIdx.x * 8 + w;

  const float4* SgP = (const float4*)(Sg + (size_t)pair * MDIM * 4);
  const float4* RP  = (const float4*)(R  + (size_t)pair * MDIM * 4);
  float4 sg = {0.f, 0.f, 0.f, 0.f}, rr = {0.f, 0.f, 0.f, 0.f};
  if (lane < MDIM) { sg = SgP[lane]; rr = RP[lane]; }
  sR4[w][lane] = rr;

  // layer0: h0 = sigmoid(Sg @ We0 + be0), lane = row m
  _Float16* H = sHa[w];
#pragma unroll
  for (int j = 0; j < NE; ++j) {
    float hv = sg.x * sWe0[0 * NE + j] + sg.y * sWe0[1 * NE + j] +
               sg.z * sWe0[2 * NE + j] + sg.w * sWe0[3 * NE + j] + sbe0[j];
    hv = sigmoid_f(hv);
    H[lane * NE + j] = (lane < MDIM) ? (_Float16)hv : (_Float16)0.0f;
  }
  __syncthreads();

  // layer1: h1 = tanh(h0 @ We1 + be1)  -> sHb   (4 WMMAs)
#pragma unroll
  for (int mh = 0; mh < 2; ++mh) {
    v16h a = load_a_frag(sHa[w] + mh * 16 * NE, lane, NE);
#pragma unroll
    for (int t = 0; t < 2; ++t) {
      float bias = sbe1[(lane & 15) + t * 16];
      v8f c;
#pragma unroll
      for (int r = 0; r < 8; ++r) c[r] = bias;
      v16h b = *(const v16h*)(sB1 + t * 512 + lane * 16);
      c = wmma16x16x32(a, b, c);
#pragma unroll
      for (int r = 0; r < 8; ++r) {
        int row = mh * 16 + r + ((lane & 16) ? 8 : 0);
        int col = (lane & 15) + t * 16;
        sHb[w][row * NE + col] = (_Float16)tanh_f(c[r]);
      }
    }
  }
  __syncthreads();

  // layer2: G = h1 @ We2 + be2 (no activation) -> sHa   (4 WMMAs)
#pragma unroll
  for (int mh = 0; mh < 2; ++mh) {
    v16h a = load_a_frag(sHb[w] + mh * 16 * NE, lane, NE);
#pragma unroll
    for (int t = 0; t < 2; ++t) {
      float bias = sbe2[(lane & 15) + t * 16];
      v8f c;
#pragma unroll
      for (int r = 0; r < 8; ++r) c[r] = bias;
      v16h b = *(const v16h*)(sB2 + t * 512 + lane * 16);
      c = wmma16x16x32(a, b, c);
#pragma unroll
      for (int r = 0; r < 8; ++r) {
        int row = mh * 16 + r + ((lane & 16) ? 8 : 0);
        int col = (lane & 15) + t * 16;
        sHa[w][row * NE + col] = (_Float16)c[r];
      }
    }
  }
  __syncthreads();

  // Contractions: Bm[l][lane] = sum_m R[m][l]*G[m][lane]; A8[k][l] per lane
  const _Float16* G = sHa[w];
  float bm0 = 0.f, bm1 = 0.f, bm2 = 0.f, bm3 = 0.f;
  int kk = lane >> 2, ll = lane & 3;
  float a8 = 0.f;
  for (int m = 0; m < MDIM; ++m) {
    float4 rm = sR4[w][m];
    float g = (float)G[m * NE + lane];
    bm0 += rm.x * g; bm1 += rm.y * g; bm2 += rm.z * g; bm3 += rm.w * g;
    float rl = (ll == 0) ? rm.x : (ll == 1) ? rm.y : (ll == 2) ? rm.z : rm.w;
    a8 += (float)G[m * NE + kk] * rl;
  }
  sA8[w][lane] = a8;
  __syncthreads();

  // D[k][j] = sum_l A8[k][l]*Bm[l][j]; store as f16 in A-fragment order.
  // D flat col c = k*32 + j  ->  k-step s = k, kk = j.
  int grp = pair >> 4, lr = pair & 15;
  int j = lane;
  int hi16 = j >> 4;
  int wlow = j & 15;
  int lane_hi = wlow >> 3;
  int rem = wlow & 7;
  int v = hi16 * 4 + (rem >> 1);
  int h = rem & 1;
  int flane = lr + lane_hi * 16;
  float bm[4] = {bm0, bm1, bm2, bm3};
#pragma unroll
  for (int k = 0; k < NE4; ++k) {
    float d = 0.f;
#pragma unroll
    for (int l = 0; l < 4; ++l) d += sA8[w][k * 4 + l] * bm[l];
    wsD[(size_t)(grp * 8 + k) * 512 + flane * 16 + v * 2 + h] = (_Float16)d;
  }
}

// ---------------------------------------------------------------------------
// k2: fit MLP on D (all GEMMs via WMMA) + per-b accumulation.
// One 16-row group per wave; 8 waves/block; 48 WMMAs per wave.
__global__ void __launch_bounds__(256)
k2_fit(const _Float16* __restrict__ wsD,
       const _Float16* __restrict__ pBf0, const _Float16* __restrict__ pBf1,
       const _Float16* __restrict__ pBf2,
       const float* __restrict__ bf0, const float* __restrict__ bf1,
       const float* __restrict__ bf2,
       const float* __restrict__ Wf3, const float* __restrict__ bf3,
       float* __restrict__ acc) {
  __shared__ float    sbf0[NF], sbf1[NF], sbf2[NF], sW3[NF];
  __shared__ _Float16 sStg[8][16 * NF];   // per-wave 16x64 f16 stage
  __shared__ float    sQ[8][16];

  int tid = threadIdx.x;
  if (tid < NF) { sbf0[tid] = bf0[tid]; sbf1[tid] = bf1[tid];
                  sbf2[tid] = bf2[tid]; sW3[tid]  = Wf3[tid]; }
  __syncthreads();

  int lane = tid & 31, w = tid >> 5;
  int gidx = blockIdx.x * 8 + w;                      // 16-row group id
  const _Float16* Arow = wsD + (size_t)gidx * 8 * 512;

  // layer0: [16,256] @ [256,64] ; A frags streamed from ws (32B/lane v16h)
  v8f c0[4];
#pragma unroll
  for (int t = 0; t < 4; ++t) {
    float b = sbf0[(lane & 15) + t * 16];
#pragma unroll
    for (int r = 0; r < 8; ++r) c0[t][r] = b;
  }
#pragma unroll
  for (int s = 0; s < 8; ++s) {
    v16h a = *(const v16h*)(Arow + s * 512 + lane * 16);
#pragma unroll
    for (int t = 0; t < 4; ++t) {
      v16h b = *(const v16h*)(pBf0 + (s * 4 + t) * 512 + lane * 16);
      c0[t] = wmma16x16x32(a, b, c0[t]);
    }
  }
#pragma unroll
  for (int t = 0; t < 4; ++t)
#pragma unroll
    for (int r = 0; r < 8; ++r) {
      int row = r + ((lane & 16) ? 8 : 0);
      int col = (lane & 15) + t * 16;
      sStg[w][row * NF + col] = (_Float16)tanh_f(c0[t][r]);
    }
  __syncthreads();

  // layer1: [16,64] @ [64,64], sigmoid
  v8f c1[4];
#pragma unroll
  for (int t = 0; t < 4; ++t) {
    float b = sbf1[(lane & 15) + t * 16];
#pragma unroll
    for (int r = 0; r < 8; ++r) c1[t][r] = b;
  }
#pragma unroll
  for (int s = 0; s < 2; ++s) {
    v16h a = load_a_frag(sStg[w] + s * 32, lane, NF);
#pragma unroll
    for (int t = 0; t < 4; ++t) {
      v16h b = *(const v16h*)(pBf1 + (s * 4 + t) * 512 + lane * 16);
      c1[t] = wmma16x16x32(a, b, c1[t]);
    }
  }
  __syncthreads();
#pragma unroll
  for (int t = 0; t < 4; ++t)
#pragma unroll
    for (int r = 0; r < 8; ++r) {
      int row = r + ((lane & 16) ? 8 : 0);
      int col = (lane & 15) + t * 16;
      sStg[w][row * NF + col] = (_Float16)sigmoid_f(c1[t][r]);
    }
  __syncthreads();

  // layer2: [16,64] @ [64,64], tanh
  v8f c2[4];
#pragma unroll
  for (int t = 0; t < 4; ++t) {
    float b = sbf2[(lane & 15) + t * 16];
#pragma unroll
    for (int r = 0; r < 8; ++r) c2[t][r] = b;
  }
#pragma unroll
  for (int s = 0; s < 2; ++s) {
    v16h a = load_a_frag(sStg[w] + s * 32, lane, NF);
#pragma unroll
    for (int t = 0; t < 4; ++t) {
      v16h b = *(const v16h*)(pBf2 + (s * 4 + t) * 512 + lane * 16);
      c2[t] = wmma16x16x32(a, b, c2[t]);
    }
  }
  __syncthreads();
#pragma unroll
  for (int t = 0; t < 4; ++t)
#pragma unroll
    for (int r = 0; r < 8; ++r) {
      int row = r + ((lane & 16) ? 8 : 0);
      int col = (lane & 15) + t * 16;
      sStg[w][row * NF + col] = (_Float16)tanh_f(c2[t][r]);
    }
  __syncthreads();

  // layer3: Q[row] = h @ Wf3 + bf3 ; lanes 0-15, row = lane
  if (lane < 16) {
    float q = bf3[0];
#pragma unroll
    for (int n = 0; n < NF; ++n) q += (float)sStg[w][lane * NF + n] * sW3[n];
    sQ[w][lane] = q;
  }
  __syncthreads();
  if (lane == 0) {
    float sum = 0.f;
#pragma unroll
    for (int i = 0; i < 16; ++i) sum += sQ[w][i];
    atomicAdd(&acc[gidx >> 5], sum);  // 32 groups of 16 rows per b
  }
}

// ---------------------------------------------------------------------------
__global__ void k3_final(const float* __restrict__ acc, float* __restrict__ out) {
  if (threadIdx.x < BDIM)
    out[threadIdx.x] = acc[threadIdx.x] * (1.0f / (float)NDIM);
}

// ---------------------------------------------------------------------------
extern "C" void kernel_launch(void* const* d_in, const int* in_sizes, int n_in,
                              void* d_out, int out_size, void* d_ws, size_t ws_size,
                              hipStream_t stream) {
  const float* Sg  = (const float*)d_in[0];
  const float* R   = (const float*)d_in[1];
  const float* We0 = (const float*)d_in[2];
  const float* be0 = (const float*)d_in[3];
  const float* We1 = (const float*)d_in[4];
  const float* be1 = (const float*)d_in[5];
  const float* We2 = (const float*)d_in[6];
  const float* be2 = (const float*)d_in[7];
  const float* Wf0 = (const float*)d_in[8];
  const float* bf0 = (const float*)d_in[9];
  const float* Wf1 = (const float*)d_in[10];
  const float* bf1 = (const float*)d_in[11];
  const float* Wf2 = (const float*)d_in[12];
  const float* bf2 = (const float*)d_in[13];
  const float* Wf3 = (const float*)d_in[14];
  const float* bf3 = (const float*)d_in[15];

  // Workspace layout
  _Float16* wsD  = (_Float16*)d_ws;                     // PAIRS*256 f16 = 16 MB
  size_t offD    = (size_t)PAIRS * 256;
  _Float16* pBf0 = wsD + offD;                          // 8*4*512 f16
  _Float16* pBf1 = pBf0 + 8 * 4 * 512;                  // 2*4*512 f16
  _Float16* pBf2 = pBf1 + 2 * 4 * 512;                  // 2*4*512 f16
  float*    acc  = (float*)(pBf2 + 2 * 4 * 512);        // 64 f32 (4B aligned)

  k0_zero<<<1, 64, 0, stream>>>(acc);
  k_pack<<<64, 256, 0, stream>>>(Wf0, Wf1, Wf2, pBf0, pBf1, pBf2);
  k1_emb<<<PAIRS / 8, 256, 0, stream>>>(Sg, R, We0, be0, We1, be1, We2, be2, wsD);
  k2_fit<<<GROUPS / 8, 256, 0, stream>>>(wsD, pBf0, pBf1, pBf2,
                                         bf0, bf1, bf2, Wf3, bf3, acc);
  k3_final<<<1, 64, 0, stream>>>(acc, (float*)d_out);
}
